// GrowingFieldV2_15152644620827
// MI455X (gfx1250) — matrix operand order repaired and checked by reading, not compiled, not run on hardware
//
#include <hip/hip_runtime.h>
#include <math.h>

// ---------------- problem constants ----------------
#define BATCH 8192
#define KIN   3072
#define NNEU  500
#define NPAD  512
#define MT    64          // batch rows per workgroup
#define OUTS  10
// LDS strides (bf16 elements), padded against 64-bank conflicts
#define XS    40          // x staging stride     (40*2B = 20 dwords -> 20i mod 64, 16 distinct)
#define WS    40          // weight staging stride
#define AS    520         // activation tile stride (520*2B = 260 dw -> 4i mod 64 rotation)
#define LDS_BYTES (2 * MT * AS * 2)   // ACTh + ACTl = 133120 B (staging overlays this)

typedef __attribute__((ext_vector_type(4)))  __bf16 v4bf;
typedef __attribute__((ext_vector_type(8)))  __bf16 v8bf;
typedef __attribute__((ext_vector_type(16))) __bf16 v16bf;
typedef __attribute__((ext_vector_type(8)))  float  v8f;

__device__ __forceinline__ float clamp_pos(float p) {
    return fminf(fmaxf(p, 0.1f), 100.0f - 0.1f);
}

struct bf2 { __bf16 h; __bf16 l; };

__device__ __forceinline__ bf2 split1(float v) {
    bf2 r;
    r.h = (__bf16)v;
    r.l = (__bf16)(v - (float)r.h);
    return r;
}

// build a 16-element bf16 fragment from two contiguous 16B chunks (p, p+off)
__device__ __forceinline__ v16bf frag2(const __bf16* p, int off) {
    union { v16bf v; v8bf h[2]; } u;
    u.h[0] = *(const v8bf*)p;
    u.h[1] = *(const v8bf*)(p + off);
    return u.v;
}

// ---------------- setup kernel 1: normalized features ----------------
__global__ void gf_featnorm(const float* __restrict__ feat, float* __restrict__ fn) {
    int row = blockIdx.x;          // 0..499
    int t   = threadIdx.x;         // 0..63
    float v = feat[row * 64 + t];
    __shared__ float s[64];
    s[t] = v * v;
    __syncthreads();
    for (int off = 32; off > 0; off >>= 1) {
        if (t < off) s[t] += s[t + off];
        __syncthreads();
    }
    float nrm = fmaxf(sqrtf(s[0]), 1e-6f);
    fn[row * 64 + t] = v / nrm;
}

// ---------------- setup kernel 2: row-normalized conn weights -> bf16 hi/lo
// grid = 512 blocks (row i), block = 512 threads (col j); rows/cols >=500 -> 0
__global__ void gf_conn(const float* __restrict__ pos,
                        const float* __restrict__ fn,
                        __bf16* __restrict__ ch, __bf16* __restrict__ cl) {
    int i = blockIdx.x;
    int j = threadIdx.x;
    __shared__ float red[512];
    float w = 0.0f;
    if (i < NNEU && j < NNEU) {
        float dx = clamp_pos(pos[i * 3 + 0]) - clamp_pos(pos[j * 3 + 0]);
        float dy = clamp_pos(pos[i * 3 + 1]) - clamp_pos(pos[j * 3 + 1]);
        float dz = clamp_pos(pos[i * 3 + 2]) - clamp_pos(pos[j * 3 + 2]);
        float sq = dx * dx + dy * dy + dz * dz;
        if (sq > 0.0f) {
            float dist = sqrtf(sq);
            if (dist < 20.0f) {
                float att = expf(-dist * (1.0f / 20.0f));
                float dot = 0.0f;
                for (int k = 0; k < 64; k++)
                    dot += fn[i * 64 + k] * fn[j * 64 + k];
                dot = fminf(fmaxf(dot, -1.0f), 1.0f);
                w = att * (0.5f + 0.5f * dot);
            }
        }
    }
    red[j] = w;
    __syncthreads();
    for (int off = 256; off > 0; off >>= 1) {
        if (j < off) red[j] += red[j + off];
        __syncthreads();
    }
    float wn = w / (red[0] + 1e-6f);
    bf2 s = split1(wn);
    ch[(size_t)i * NPAD + j] = s.h;
    cl[(size_t)i * NPAD + j] = s.l;
}

// ---------------- setup kernel 3: split input weights -> bf16 hi/lo --------
// grid = 512 blocks (neuron row, zero pad >=500), block = 256 threads
__global__ void gf_splitw(const float* __restrict__ win,
                          __bf16* __restrict__ wh, __bf16* __restrict__ wl) {
    int row = blockIdx.x;
    int t   = threadIdx.x;
    #pragma unroll
    for (int i = 0; i < 3; i++) {
        int c = (t + i * 256) << 2;            // 0..3068 step 4
        float4 v = make_float4(0.f, 0.f, 0.f, 0.f);
        if (row < NNEU) v = *(const float4*)&win[(size_t)row * KIN + c];
        v4bf h, l;
        bf2 s0 = split1(v.x); h[0] = s0.h; l[0] = s0.l;
        bf2 s1 = split1(v.y); h[1] = s1.h; l[1] = s1.l;
        bf2 s2 = split1(v.z); h[2] = s2.h; l[2] = s2.l;
        bf2 s3 = split1(v.w); h[3] = s3.h; l[3] = s3.l;
        *(v4bf*)&wh[(size_t)row * KIN + c] = h;
        *(v4bf*)&wl[(size_t)row * KIN + c] = l;
    }
}

// ---------------- setup kernel 4: gating vectors -------------------------
__global__ void gf_gate(const float* __restrict__ pos,
                        const float* __restrict__ biases,
                        const float* __restrict__ wout,
                        float* __restrict__ in_w,
                        float* __restrict__ bias_p,
                        float* __restrict__ wout_s) {
    int t = threadIdx.x;
    __shared__ float sa[512];
    __shared__ float sb[512];
    float ein = 0.0f, eout = 0.0f;
    if (t < NNEU) {
        float xn = clamp_pos(pos[t * 3 + 0]) * (1.0f / 100.0f);
        ein  = expf(-2.0f * xn);
        eout = expf(2.0f * (xn - 1.0f));
    }
    sa[t] = ein;
    sb[t] = eout;
    __syncthreads();
    for (int off = 256; off > 0; off >>= 1) {
        if (t < off) { sa[t] += sa[t + off]; sb[t] += sb[t + off]; }
        __syncthreads();
    }
    float si = sa[0], so = sb[0];
    in_w[t]   = (t < NNEU) ? ein / (si + 1e-6f) : 0.0f;
    bias_p[t] = (t < NNEU) ? biases[t] : 0.0f;
    float ow  = (t < NNEU) ? eout / (so + 1e-6f) : 0.0f;
    for (int o = 0; o < OUTS; o++)
        wout_s[t * OUTS + o] = (t < NNEU) ? wout[t * OUTS + o] * ow : 0.0f;
}

// ---------------- fused main kernel: bf16x3 split-precision WMMA ----------
// block = 256 threads (8 waves), tile = 64 batch rows x 512 neurons
__launch_bounds__(256)
__global__ void gf_field(const float*  __restrict__ x,
                         const __bf16* __restrict__ winh,
                         const __bf16* __restrict__ winl,
                         const __bf16* __restrict__ ch,
                         const __bf16* __restrict__ cl,
                         const float*  __restrict__ in_w,
                         const float*  __restrict__ bias_p,
                         const float*  __restrict__ wout_s,
                         float* __restrict__ out) {
    extern __shared__ char smemc[];
    // persistent (post-GEMM) activation tile, hi/lo bf16
    __bf16* ACTh = (__bf16*)smemc;           // [MT][AS]
    __bf16* ACTl = ACTh + MT * AS;           // [MT][AS]
    // GEMM staging overlays the same LDS
    __bf16* Xh = (__bf16*)smemc;             // [MT][XS]
    __bf16* Xl = Xh + MT * XS;
    __bf16* Wh = Xl + MT * XS;               // [NPAD][WS]
    __bf16* Wl = Wh + NPAD * WS;

    const int tid  = threadIdx.x;
    const int lane = tid & 31;
    const int wv   = tid >> 5;
    const int lm   = lane & 15;
    const int ca   = (lane < 16) ? 0 : 8;    // A-frag K base (16-bit layout)
    const int kb   = (lane < 16) ? 0 : 8;    // B-frag K base
    const int mrow = (lane < 16) ? 0 : 8;    // C/D row offset per lane half
    const int n0   = wv * 64;
    const int m0   = blockIdx.x * MT;

    v8f acc[4][4];
    #pragma unroll
    for (int mi = 0; mi < 4; mi++)
        #pragma unroll
        for (int ni = 0; ni < 4; ni++)
            #pragma unroll
            for (int r = 0; r < 8; r++) acc[mi][ni][r] = 0.0f;

    // ---------------- big GEMM: act = x @ win.T (bf16x3) -----------------
    for (int kk = 0; kk < KIN; kk += 32) {
        __syncthreads();
        // stage x chunk 64x32: fp32 -> bf16 hi/lo (split inline, read once)
        #pragma unroll
        for (int i = 0; i < 2; i++) {
            int idx = tid + i * 256;
            int row = idx >> 3, c = (idx & 7) << 2;
            float4 v = *(const float4*)&x[(size_t)(m0 + row) * KIN + kk + c];
            v4bf h, l;
            bf2 s0 = split1(v.x); h[0] = s0.h; l[0] = s0.l;
            bf2 s1 = split1(v.y); h[1] = s1.h; l[1] = s1.l;
            bf2 s2 = split1(v.z); h[2] = s2.h; l[2] = s2.l;
            bf2 s3 = split1(v.w); h[3] = s3.h; l[3] = s3.l;
            *(v4bf*)&Xh[row * XS + c] = h;
            *(v4bf*)&Xl[row * XS + c] = l;
        }
        // stage weight chunk 512x32: pre-split bf16, straight copy
        #pragma unroll
        for (int i = 0; i < 8; i++) {
            int idx = tid + i * 256;
            int row = idx >> 2, c = (idx & 3) << 3;
            *(v8bf*)&Wh[row * WS + c] = *(const v8bf*)&winh[(size_t)row * KIN + kk + c];
            *(v8bf*)&Wl[row * WS + c] = *(const v8bf*)&winl[(size_t)row * KIN + kk + c];
        }
        __syncthreads();

        v16bf ah[4], al[4], bh[4], bl[4];
        #pragma unroll
        for (int mi = 0; mi < 4; mi++) {
            ah[mi] = frag2(&Xh[(mi * 16 + lm) * XS + ca], 16);
            al[mi] = frag2(&Xl[(mi * 16 + lm) * XS + ca], 16);
        }
        #pragma unroll
        for (int ni = 0; ni < 4; ni++) {
            bh[ni] = frag2(&Wh[(n0 + ni * 16 + lm) * WS + kb], 16);
            bl[ni] = frag2(&Wl[(n0 + ni * 16 + lm) * WS + kb], 16);
        }
        #pragma unroll
        for (int mi = 0; mi < 4; mi++)
            #pragma unroll
            for (int ni = 0; ni < 4; ni++) {
                acc[mi][ni] = __builtin_amdgcn_wmma_f32_16x16x32_bf16(
                    false, ah[mi], false, bh[ni], (short)0, acc[mi][ni], false, false);
                acc[mi][ni] = __builtin_amdgcn_wmma_f32_16x16x32_bf16(
                    false, ah[mi], false, bl[ni], (short)0, acc[mi][ni], false, false);
                acc[mi][ni] = __builtin_amdgcn_wmma_f32_16x16x32_bf16(
                    false, al[mi], false, bh[ni], (short)0, acc[mi][ni], false, false);
            }
    }

    // input gating + bias (per-neuron, uniform across the row dim)
    #pragma unroll
    for (int ni = 0; ni < 4; ni++) {
        int n = n0 + ni * 16 + lm;
        float wi = in_w[n];
        float bi = bias_p[n];
        #pragma unroll
        for (int mi = 0; mi < 4; mi++)
            #pragma unroll
            for (int r = 0; r < 8; r++)
                acc[mi][ni][r] = acc[mi][ni][r] * wi + bi;
    }

    // spill activation tile into LDS as split bf16 (reuses staging space)
    __syncthreads();
    #pragma unroll
    for (int mi = 0; mi < 4; mi++)
        #pragma unroll
        for (int ni = 0; ni < 4; ni++) {
            int n = n0 + ni * 16 + lm;
            #pragma unroll
            for (int r = 0; r < 8; r++) {
                bf2 s = split1(acc[mi][ni][r]);
                int m = mi * 16 + r + mrow;
                ACTh[m * AS + n] = s.h;
                ACTl[m * AS + n] = s.l;
            }
        }
    __syncthreads();

    // ---------------- 3 message-passing iterations (bf16x3) --------------
    for (int it = 0; it < 3; it++) {
        v8f macc[4][4];
        #pragma unroll
        for (int mi = 0; mi < 4; mi++)
            #pragma unroll
            for (int ni = 0; ni < 4; ni++)
                #pragma unroll
                for (int r = 0; r < 8; r++) macc[mi][ni][r] = 0.0f;

        // msg[m,i] = sum_j act[m,j] * conn_w[i,j]
        #pragma unroll 2
        for (int j = 0; j < NPAD; j += 32) {
            v16bf ah[4], al[4], bh[4], bl[4];
            #pragma unroll
            for (int mi = 0; mi < 4; mi++) {
                ah[mi] = frag2(&ACTh[(mi * 16 + lm) * AS + j + ca], 16);
                al[mi] = frag2(&ACTl[(mi * 16 + lm) * AS + j + ca], 16);
            }
            #pragma unroll
            for (int ni = 0; ni < 4; ni++) {
                size_t nb = (size_t)(n0 + ni * 16 + lm) * NPAD + j + kb;
                bh[ni] = frag2(&ch[nb], 16);
                bl[ni] = frag2(&cl[nb], 16);
            }
            #pragma unroll
            for (int mi = 0; mi < 4; mi++)
                #pragma unroll
                for (int ni = 0; ni < 4; ni++) {
                    macc[mi][ni] = __builtin_amdgcn_wmma_f32_16x16x32_bf16(
                        false, ah[mi], false, bh[ni], (short)0, macc[mi][ni], false, false);
                    macc[mi][ni] = __builtin_amdgcn_wmma_f32_16x16x32_bf16(
                        false, ah[mi], false, bl[ni], (short)0, macc[mi][ni], false, false);
                    macc[mi][ni] = __builtin_amdgcn_wmma_f32_16x16x32_bf16(
                        false, al[mi], false, bh[ni], (short)0, macc[mi][ni], false, false);
                }
        }

        // act = min(relu(act + 0.5*msg), 50)
        #pragma unroll
        for (int ni = 0; ni < 4; ni++) {
            int n = n0 + ni * 16 + lm;
            #pragma unroll
            for (int mi = 0; mi < 4; mi++)
                #pragma unroll
                for (int r = 0; r < 8; r++) {
                    int m = mi * 16 + r + mrow;
                    float old = (float)ACTh[m * AS + n] + (float)ACTl[m * AS + n];
                    float v = old + 0.5f * macc[mi][ni][r];
                    macc[mi][ni][r] = fminf(fmaxf(v, 0.0f), 50.0f);
                }
        }
        __syncthreads();
        #pragma unroll
        for (int ni = 0; ni < 4; ni++) {
            int n = n0 + ni * 16 + lm;
            #pragma unroll
            for (int mi = 0; mi < 4; mi++)
                #pragma unroll
                for (int r = 0; r < 8; r++) {
                    bf2 s = split1(macc[mi][ni][r]);
                    int m = mi * 16 + r + mrow;
                    ACTh[m * AS + n] = s.h;
                    ACTl[m * AS + n] = s.l;
                }
        }
        __syncthreads();
    }

    // ---------------- output projection (tiny) ---------------------------
    for (int idx = tid; idx < MT * OUTS; idx += 256) {
        int m = idx / OUTS;
        int o = idx - m * OUTS;
        float s = 0.0f;
        for (int n = 0; n < NNEU; n++) {
            float a = (float)ACTh[m * AS + n] + (float)ACTl[m * AS + n];
            s += a * wout_s[n * OUTS + o];
        }
        out[(size_t)(m0 + m) * OUTS + o] = s;
    }
}

// ---------------- launcher ------------------------------------------------
extern "C" void kernel_launch(void* const* d_in, const int* in_sizes, int n_in,
                              void* d_out, int out_size, void* d_ws, size_t ws_size,
                              hipStream_t stream) {
    const float* x         = (const float*)d_in[0];   // [8192,3072]
    const float* positions = (const float*)d_in[1];   // [500,3]
    const float* win       = (const float*)d_in[2];   // [500,3072]
    const float* features  = (const float*)d_in[3];   // [500,64]
    const float* wout      = (const float*)d_in[4];   // [500,10]
    const float* biases    = (const float*)d_in[5];   // [500]
    float* out = (float*)d_out;                       // [8192,10]

    char* wsb = (char*)d_ws;
    __bf16* winh  = (__bf16*)(wsb + 0);          // 512*3072*2 = 3145728 B
    __bf16* winl  = (__bf16*)(wsb + 3145728);    // 3145728 B
    __bf16* ch    = (__bf16*)(wsb + 6291456);    // 512*512*2  = 524288 B
    __bf16* cl    = (__bf16*)(wsb + 6815744);    // 524288 B
    float*  in_w  = (float*) (wsb + 7340032);    // 2048 B
    float*  biasp = (float*) (wsb + 7342080);    // 2048 B
    float*  wouts = (float*) (wsb + 7344128);    // 20480 B
    float*  fnorm = (float*) (wsb + 7364608);    // 128000 B

    gf_featnorm<<<NNEU, 64, 0, stream>>>(features, fnorm);
    gf_conn<<<NPAD, NPAD, 0, stream>>>(positions, fnorm, ch, cl);
    gf_splitw<<<NPAD, 256, 0, stream>>>(win, winh, winl);
    gf_gate<<<1, NPAD, 0, stream>>>(positions, biases, wout, in_w, biasp, wouts);
    gf_field<<<BATCH / MT, 256, LDS_BYTES, stream>>>(
        x, winh, winl, ch, cl, in_w, biasp, wouts, out);
}
